// jgMPNN_3616362463932
// MI455X (gfx1250) — matrix operand order repaired and checked by reading, not compiled, not run on hardware
//
#include <hip/hip_runtime.h>

#define NN 2048
#define HIDK 5

typedef __bf16 bf16t;
typedef bf16t v16bf __attribute__((ext_vector_type(16)));
typedef float v8f  __attribute__((ext_vector_type(8)));

__device__ __forceinline__ unsigned short f2bf(float x) {
    union { float f; unsigned int u; } v; v.f = x;
    unsigned int r = v.u + 0x7FFFu + ((v.u >> 16) & 1u);   // round-to-nearest-even
    return (unsigned short)(r >> 16);
}

// ---------------------------------------------------------------------------
// Pack: a -> Ab (bf16, row-major) and At (bf16, transposed); f -> Ft (bf16,
// transposed). 64x64 tile per block, LDS transpose for coalesced stores.
// ---------------------------------------------------------------------------
__global__ void pack_kernel(const float* __restrict__ a, const float* __restrict__ f,
                            unsigned short* __restrict__ Ab,
                            unsigned short* __restrict__ At,
                            unsigned short* __restrict__ Ft) {
    __shared__ __align__(16) unsigned short L[64][72];
    const int tid = threadIdx.x;
    const int bi = blockIdx.y * 64;   // row base
    const int bj = blockIdx.x * 64;   // col base

    // --- a -> Ab + At ---
    #pragma unroll
    for (int p = 0; p < 16; ++p) {
        int t = tid + p * 256;
        int r = t >> 6, c = t & 63;
        unsigned short us = f2bf(a[(size_t)(bi + r) * NN + (bj + c)]);
        Ab[(size_t)(bi + r) * NN + (bj + c)] = us;
        L[c][r] = us;
    }
    __syncthreads();
    #pragma unroll
    for (int p = 0; p < 16; ++p) {
        int t = tid + p * 256;
        int r = t >> 6, c = t & 63;
        At[(size_t)(bj + r) * NN + (bi + c)] = L[r][c];
    }
    __syncthreads();
    // --- f -> Ft ---
    #pragma unroll
    for (int p = 0; p < 16; ++p) {
        int t = tid + p * 256;
        int r = t >> 6, c = t & 63;
        L[c][r] = f2bf(f[(size_t)(bi + r) * NN + (bj + c)]);
    }
    __syncthreads();
    #pragma unroll
    for (int p = 0; p < 16; ++p) {
        int t = tid + p * 256;
        int r = t >> 6, c = t & 63;
        Ft[(size_t)(bj + r) * NN + (bi + c)] = L[r][c];
    }
}

// ---------------------------------------------------------------------------
// Elementwise: aggn = (raw + raw^T) / (2*de2); mini-MLP over HID channels;
// G = sum_k W2[HID+k]*relu(...) (bf16), hsum = sum_k W2[k]*relu(...) -> d_out.
// ---------------------------------------------------------------------------
__global__ void edge_mlp_kernel(const float* __restrict__ f, const float* __restrict__ de2,
                                const float* __restrict__ raw,
                                const float* __restrict__ W1, const float* __restrict__ b1,
                                const float* __restrict__ W2,
                                unsigned short* __restrict__ G, float* __restrict__ hsum) {
    const int idx = blockIdx.x * blockDim.x + threadIdx.x;  // [0, N*N)
    const int i = idx >> 11;
    const int j = idx & (NN - 1);
    const float fij  = f[idx];
    const float aggn = (raw[idx] + raw[(size_t)j * NN + i]) / (2.0f * de2[idx]);
    float g = 0.f, hs = 0.f;
    #pragma unroll
    for (int k = 0; k < HIDK; ++k) {
        float t = fmaf(fij, W1[2 * k], fmaf(aggn, W1[2 * k + 1], b1[k]));
        t = fmaxf(t, 0.0f);
        hs = fmaf(W2[k], t, hs);
        g  = fmaf(W2[HIDK + k], t, g);
    }
    G[idx]    = f2bf(g);
    hsum[idx] = hs;
}

// ---------------------------------------------------------------------------
// TN bf16 GEMM: C[m,n] = sum_ph sum_k X_ph[m,k] * Y_ph[n,k]  (f32 accumulate)
// Block tile 128x128x64, 4 wave32 waves (2x2), each wave 64x64 = 16 WMMA tiles.
// Global->LDS via GLOBAL_LOAD_ASYNC_TO_LDS_B128 (ASYNCcnt), double-buffered,
// single barrier per K-tile, phases flattened into one pipelined loop.
// EPI: C = b2 + hsum + C/de2 (hsum may alias C).
// ---------------------------------------------------------------------------
#define BM 128
#define BN 128
#define BK 64
#define BKP 88   // row stride in ushorts: 176B = 11*16B -> aligned b128, low conflicts
#define KIT (NN / BK)

union Frag { uint4 q[2]; v16bf v; };

__device__ __forceinline__ unsigned lds_off(const void* p) {
    // flat LDS addressing uses addr[31:0] as the LDS offset (ISA 10.2)
    return (unsigned)(size_t)p;
}

__device__ __forceinline__ void load_tile_async(unsigned short (*Xs)[BKP],
                                                unsigned short (*Ys)[BKP],
                                                const unsigned short* X,
                                                const unsigned short* Y,
                                                int mBase, int nBase, int k0, int tid) {
    const int r0 = tid >> 3;          // 0..15
    const int c  = (tid & 7) * 8;     // ushort col, 16B granules
    #pragma unroll
    for (int p = 0; p < 8; ++p) {
        const int row = r0 + p * 16;
        const unsigned short* gx = &X[(size_t)(mBase + row) * NN + k0 + c];
        const unsigned short* gy = &Y[(size_t)(nBase + row) * NN + k0 + c];
        const unsigned lx = lds_off(&Xs[row][c]);
        const unsigned ly = lds_off(&Ys[row][c]);
        asm volatile("global_load_async_to_lds_b128 %0, %1, off"
                     :: "v"(lx), "v"(gx) : "memory");
        asm volatile("global_load_async_to_lds_b128 %0, %1, off"
                     :: "v"(ly), "v"(gy) : "memory");
    }
}

__device__ __forceinline__ void wait_async_all() {
    asm volatile("s_wait_asynccnt 0x0" ::: "memory");
}

template <int PHASES, bool EPI>
__global__ __launch_bounds__(128)
void gemm_tn(const unsigned short* __restrict__ X0, const unsigned short* __restrict__ Y0,
             const unsigned short* __restrict__ X1, const unsigned short* __restrict__ Y1,
             float* C,
             const float* __restrict__ de2, const float* hsum,
             const float* __restrict__ b2) {
    __shared__ __align__(16) unsigned short Xs[2][BM][BKP];
    __shared__ __align__(16) unsigned short Ys[2][BN][BKP];
    const int tid  = threadIdx.x;
    const int lane = tid & 31;
    const int wave = tid >> 5;
    const int wr   = wave & 1;    // 2 waves along M (64 rows each)
    const int wc   = wave >> 1;   // 2 waves along N (64 cols each)
    const int half = lane >> 4;
    const int l16  = lane & 15;
    const int mBase = blockIdx.y * BM;
    const int nBase = blockIdx.x * BN;

    v8f acc[4][4] = {};

    const int iters = PHASES * KIT;
    // prologue: stage tile 0 into buffer 0
    load_tile_async(Xs[0], Ys[0], X0, Y0, mBase, nBase, 0, tid);
    wait_async_all();
    __syncthreads();

    int cur = 0;
    for (int it = 0; it < iters; ++it) {
        // ---- prefetch next K-tile into the other buffer (async DMA to LDS) ----
        if (it + 1 < iters) {
            const int nit = it + 1;
            const unsigned short* Xn = (nit < KIT) ? X0 : X1;
            const unsigned short* Yn = (nit < KIT) ? Y0 : Y1;
            load_tile_async(Xs[cur ^ 1], Ys[cur ^ 1], Xn, Yn, mBase, nBase,
                            (nit & (KIT - 1)) * BK, tid);
        }
        // ---- compute from current buffer: 2 k-fragments of 32 per BK ----
        #pragma unroll
        for (int kf = 0; kf < BK; kf += 32) {
            Frag af[4], bf[4];
            #pragma unroll
            for (int tm = 0; tm < 4; ++tm) {
                const int row = wr * 64 + tm * 16 + l16;
                // ISA 16-bit A 16x32 layout: VGPR0-3 <- K=half*8+0..7, VGPR4-7 <- K=16+half*8+0..7
                af[tm].q[0] = *(const uint4*)&Xs[cur][row][kf + half * 8];
                af[tm].q[1] = *(const uint4*)&Xs[cur][row][kf + 16 + half * 8];
            }
            #pragma unroll
            for (int tn = 0; tn < 4; ++tn) {
                const int row = wc * 64 + tn * 16 + l16;
                // B 32x16 layout: lane = column, VGPR0-3 <- K=half*8+0..7, VGPR4-7 <- K=16+half*8+0..7
                bf[tn].q[0] = *(const uint4*)&Ys[cur][row][kf + half * 8];
                bf[tn].q[1] = *(const uint4*)&Ys[cur][row][kf + 16 + half * 8];
            }
            #pragma unroll
            for (int tm = 0; tm < 4; ++tm)
                #pragma unroll
                for (int tn = 0; tn < 4; ++tn)
                    acc[tm][tn] = __builtin_amdgcn_wmma_f32_16x16x32_bf16(
                        false, af[tm].v, false, bf[tn].v,
                        (short)0, acc[tm][tn], false, false);
        }
        // this wave's async writes to the other buffer must land before the
        // block-wide barrier publishes it
        wait_async_all();
        __syncthreads();
        cur ^= 1;
    }

    // ---- epilogue (C/D layout: VGPR r -> M=r(+8 for high half), N=l16) ----
    const float b2v = EPI ? b2[0] : 0.0f;
    #pragma unroll
    for (int tm = 0; tm < 4; ++tm) {
        #pragma unroll
        for (int tn = 0; tn < 4; ++tn) {
            const int gcol = nBase + wc * 64 + tn * 16 + l16;
            #pragma unroll
            for (int r = 0; r < 8; ++r) {
                const int grow = mBase + wr * 64 + tm * 16 + r + half * 8;
                const size_t off = (size_t)grow * NN + gcol;
                const float v = acc[tm][tn][r];
                if (EPI) {
                    C[off] = b2v + hsum[off] + v / de2[off];
                } else {
                    C[off] = v;
                }
            }
        }
    }
}

// ---------------------------------------------------------------------------
extern "C" void kernel_launch(void* const* d_in, const int* in_sizes, int n_in,
                              void* d_out, int out_size, void* d_ws, size_t ws_size,
                              hipStream_t stream) {
    const float* a   = (const float*)d_in[0];
    // d_in[1] = num_nodes (N = 2048, fixed by the reference)
    const float* f   = (const float*)d_in[2];
    const float* de2 = (const float*)d_in[3];
    const float* W1  = (const float*)d_in[4];
    const float* b1  = (const float*)d_in[5];
    const float* W2  = (const float*)d_in[6];
    const float* b2  = (const float*)d_in[7];
    float* out = (float*)d_out;

    const size_t NSQ = (size_t)NN * NN;
    char* ws = (char*)d_ws;
    unsigned short* Ab = (unsigned short*)(ws);              //  8 MB  a (bf16)
    unsigned short* At = (unsigned short*)(ws + 2 * NSQ);    //  8 MB  a^T (bf16)
    unsigned short* Ft = (unsigned short*)(ws + 4 * NSQ);    //  8 MB  f^T (bf16)
    unsigned short* G  = (unsigned short*)(ws + 6 * NSQ);    //  8 MB  G (bf16)
    float*          raw = (float*)(ws + 8 * NSQ);            // 16 MB  a^T @ f (f32)

    // 1) pack/convert/transposes
    dim3 packGrid(NN / 64, NN / 64);
    pack_kernel<<<packGrid, 256, 0, stream>>>(a, f, Ab, At, Ft);

    // 2) raw = At . Ft^T  ( = a^T @ f )
    dim3 gGrid(NN / BN, NN / BM);
    gemm_tn<1, false><<<gGrid, 128, 0, stream>>>(At, Ft, At, Ft, raw,
                                                 nullptr, nullptr, nullptr);

    // 3) elementwise MLP fold: G (bf16) and hsum -> d_out
    edge_mlp_kernel<<<(unsigned)(NSQ / 256), 256, 0, stream>>>(f, de2, raw, W1, b1, W2,
                                                               G, out);

    // 4) UV = At.G^T + G.Ab^T, fused: out = b2 + hsum + UV/de2  (hsum aliases out)
    gemm_tn<2, true><<<gGrid, 128, 0, stream>>>(At, G, G, Ab, out, de2, out, b2);
}
